// TransformerLM_223338299737
// MI455X (gfx1250) — matrix operand order repaired and checked by reading, not compiled, not run on hardware
//
#include <hip/hip_runtime.h>
#include <hip/hip_bf16.h>
#include <math.h>

// ---------------------------------------------------------------------------
// Model constants (from reference): V=32000 D=1024 L=4 H=16 T=1024 DH=64 FF=4096
// B=2 -> M = B*T = 2048 rows for all token-parallel GEMMs.
// ---------------------------------------------------------------------------
#define VV   32000
#define DD   1024
#define LLAY 4
#define HH   16
#define TT   1024
#define DHD  64
#define FFD  4096
#define BB   2
#define MM   (BB * TT)
#define SCALE_ATTN 0.125f   // DH^-0.5 = 1/8

typedef __attribute__((ext_vector_type(16))) __bf16 v16bf;
typedef __attribute__((ext_vector_type(8)))  float  v8f;

// ---------------------------------------------------------------------------
// bf16 WMMA GEMM:  C[M,N] = act( A[M,K] * W[N,K]^T + bias[N] ) (+ residual)
// Block: 256 threads = 8 waves. Tile: 128(M) x 128(N), K-step 32.
// Wave wv: wm_ = wv>>1 -> which 32-row pair, wn = wv&1 -> 64-col half.
// Each wave computes a 32x64 output patch: acc[2][4] f32 16x16 tiles.
// Software pipeline: global loads for step k+1 are issued before the WMMA
// compute of step k (consumed at the next LDS store), and step k+2 is
// prefetched with global_prefetch_b8.
// A/W are fp32 in HBM; converted to bf16 while staging to LDS (HBM bandwidth
// is ample at 23.3 TB/s; inline conversion avoids an extra conversion pass).
// Fragment layouts used here (A/B in contiguous K order, not the ISA table order):
//   A (16x32 bf16): lane l: m=l%16, hi=l/16; elem j   -> K=16*hi+j
//   B (32x16 bf16): lane l: n=l%16, hi=l/16; elem j   -> K=16*hi+j
//   C/D 16x16 f32:  lane l: n=l%16; vgpr r -> m = 8*(l/16)+r
// ---------------------------------------------------------------------------
template <int ACT> // 0 = none, 1 = exact GELU, 2 = sigmoid
__global__ __launch_bounds__(256) void gemm_bf16_wmma(
    const float* __restrict__ A, int lda,
    const float* __restrict__ W, int ldw,
    const float* __restrict__ bias,
    const float* __restrict__ residual,
    float* __restrict__ C, int ldc, int K) {
  __shared__ __bf16 As[128][40];
  __shared__ __bf16 Bs[128][40];

  const int tid  = threadIdx.x;
  const int m0   = blockIdx.y * 128;
  const int n0   = blockIdx.x * 128;
  const int lane = tid & 31;
  const int wv   = tid >> 5;
  const int wm_  = wv >> 1;   // 0..3 -> rows [wm_*32, wm_*32+32)
  const int wn   = wv & 1;    // 0..1 -> cols [wn*64, wn*64+64)
  const int lm   = lane & 15;
  const int hi   = lane >> 4;

  v8f acc[2][4] = {};

  // Staging: each thread moves 16 floats of A and 16 floats of W per K-step.
  const int sr = tid >> 1, sc = (tid & 1) * 16;

  const float* Abase = A + (size_t)(m0 + sr) * lda + sc;
  const float* Wbase = W + (size_t)(n0 + sr) * ldw + sc;

  float ra[16], rb[16];

#define LOAD_TILES(k0_)                                                       \
  {                                                                           \
    const float4* ap = reinterpret_cast<const float4*>(Abase + (k0_));        \
    float4 a0 = ap[0], a1 = ap[1], a2 = ap[2], a3 = ap[3];                    \
    ra[0] = a0.x; ra[1] = a0.y; ra[2] = a0.z; ra[3] = a0.w;                   \
    ra[4] = a1.x; ra[5] = a1.y; ra[6] = a1.z; ra[7] = a1.w;                   \
    ra[8] = a2.x; ra[9] = a2.y; ra[10] = a2.z; ra[11] = a2.w;                 \
    ra[12] = a3.x; ra[13] = a3.y; ra[14] = a3.z; ra[15] = a3.w;               \
    const float4* bp = reinterpret_cast<const float4*>(Wbase + (k0_));        \
    float4 b0 = bp[0], b1 = bp[1], b2 = bp[2], b3 = bp[3];                    \
    rb[0] = b0.x; rb[1] = b0.y; rb[2] = b0.z; rb[3] = b0.w;                   \
    rb[4] = b1.x; rb[5] = b1.y; rb[6] = b1.z; rb[7] = b1.w;                   \
    rb[8] = b2.x; rb[9] = b2.y; rb[10] = b2.z; rb[11] = b2.w;                 \
    rb[12] = b3.x; rb[13] = b3.y; rb[14] = b3.z; rb[15] = b3.w;               \
  }

  LOAD_TILES(0);

  for (int k0 = 0; k0 < K; k0 += 32) {
    // Commit the pipelined registers for this K-step to LDS (fp32 -> bf16).
#pragma unroll
    for (int j = 0; j < 16; ++j) As[sr][sc + j] = (__bf16)ra[j];
#pragma unroll
    for (int j = 0; j < 16; ++j) Bs[sr][sc + j] = (__bf16)rb[j];
    __syncthreads();

    // Issue next K-step's global loads before computing (latency overlap),
    // and prefetch the K-step after that into cache.
    if (k0 + 32 < K) {
      LOAD_TILES(k0 + 32);
      if (k0 + 64 < K) {
        __builtin_prefetch(Abase + k0 + 64, 0, 3);
        __builtin_prefetch(Wbase + k0 + 64, 0, 3);
      }
    }

    v16bf af[2];
#pragma unroll
    for (int mi = 0; mi < 2; ++mi) {
      const int arow = (wm_ * 2 + mi) * 16 + lm;
#pragma unroll
      for (int j = 0; j < 16; ++j) af[mi][j] = As[arow][16 * hi + j];
    }
#pragma unroll
    for (int nn = 0; nn < 4; ++nn) {
      v16bf bf_;
      const int bcol = wn * 64 + nn * 16 + lm;
#pragma unroll
      for (int j = 0; j < 16; ++j) bf_[j] = Bs[bcol][16 * hi + j];
#pragma unroll
      for (int mi = 0; mi < 2; ++mi) {
        acc[mi][nn] = __builtin_amdgcn_wmma_f32_16x16x32_bf16(
            false, af[mi], false, bf_, (short)0, acc[mi][nn], false, false);
      }
    }
    __syncthreads();
  }
#undef LOAD_TILES

#pragma unroll
  for (int nn = 0; nn < 4; ++nn) {
    const int col = n0 + wn * 64 + nn * 16 + lm;
    const float bv = bias ? bias[col] : 0.0f;
#pragma unroll
    for (int mi = 0; mi < 2; ++mi) {
#pragma unroll
      for (int r = 0; r < 8; ++r) {
        const int row = m0 + (wm_ * 2 + mi) * 16 + hi * 8 + r;
        float v = acc[mi][nn][r] + bv;
        if (ACT == 1) v = 0.5f * v * (1.0f + erff(v * 0.70710678118654752f));
        else if (ACT == 2) v = 1.0f / (1.0f + __expf(-v));
        const size_t idx = (size_t)row * ldc + col;
        if (residual) v += residual[idx];
        C[idx] = v;
      }
    }
  }
}

// ---------------------------------------------------------------------------
// Attention scores: S[bh,q,k] = SCALE * dot(Q[b,q,h,:], K[b,k,h,:]), DH=64.
// qkv layout: [M, 3*D], col = s*1024 + h*64 + d.
// Block: 128 threads = 4 waves; tile 64(q) x 64(k); each wave 16 q-rows x 64 k.
// Blocks fully above the causal diagonal are skipped (softmax zeroes k>q).
// ---------------------------------------------------------------------------
__global__ __launch_bounds__(128) void attn_scores(
    const float* __restrict__ qkv, float* __restrict__ scores) {
  const int q0 = blockIdx.y * 64;
  const int k0 = blockIdx.x * 64;
  if (k0 > q0 + 63) return;  // fully masked tile
  const int bh = blockIdx.z;
  const int b = bh / HH, h = bh % HH;

  __shared__ __bf16 Qs[64][40];
  __shared__ __bf16 Ks[64][40];

  const int tid = threadIdx.x;
  const int lane = tid & 31, wv = tid >> 5;
  const int lm = lane & 15, hi = lane >> 4;
  const int rr = tid >> 1, cb = (tid & 1) * 16;

  v8f acc[4] = {};

  for (int d0 = 0; d0 < DHD; d0 += 32) {
    const float4* qp = reinterpret_cast<const float4*>(
        qkv + (size_t)(b * TT + q0 + rr) * (3 * DD) + h * 64 + d0 + cb);
    const float4* kp = reinterpret_cast<const float4*>(
        qkv + (size_t)(b * TT + k0 + rr) * (3 * DD) + DD + h * 64 + d0 + cb);
    float4 q0v = qp[0], q1v = qp[1], q2v = qp[2], q3v = qp[3];
    float4 k0v = kp[0], k1v = kp[1], k2v = kp[2], k3v = kp[3];
    float qt[16] = {q0v.x, q0v.y, q0v.z, q0v.w, q1v.x, q1v.y, q1v.z, q1v.w,
                    q2v.x, q2v.y, q2v.z, q2v.w, q3v.x, q3v.y, q3v.z, q3v.w};
    float kt[16] = {k0v.x, k0v.y, k0v.z, k0v.w, k1v.x, k1v.y, k1v.z, k1v.w,
                    k2v.x, k2v.y, k2v.z, k2v.w, k3v.x, k3v.y, k3v.z, k3v.w};
#pragma unroll
    for (int j = 0; j < 16; ++j) Qs[rr][cb + j] = (__bf16)qt[j];
#pragma unroll
    for (int j = 0; j < 16; ++j) Ks[rr][cb + j] = (__bf16)kt[j];
    __syncthreads();

    v16bf af;
    const int arow = wv * 16 + lm;
#pragma unroll
    for (int j = 0; j < 16; ++j) af[j] = Qs[arow][16 * hi + j];
#pragma unroll
    for (int nn = 0; nn < 4; ++nn) {
      v16bf bf_;
      const int bcol = nn * 16 + lm;
#pragma unroll
      for (int j = 0; j < 16; ++j) bf_[j] = Ks[bcol][16 * hi + j];
      acc[nn] = __builtin_amdgcn_wmma_f32_16x16x32_bf16(
          false, af, false, bf_, (short)0, acc[nn], false, false);
    }
    __syncthreads();
  }

#pragma unroll
  for (int nn = 0; nn < 4; ++nn) {
    const int k = k0 + nn * 16 + lm;
#pragma unroll
    for (int r = 0; r < 8; ++r) {
      const int q = q0 + wv * 16 + hi * 8 + r;
      scores[((size_t)bh * TT + q) * TT + k] = SCALE_ATTN * acc[nn][r];
    }
  }
}

// ---------------------------------------------------------------------------
// Causal softmax over one row of scores (k <= q valid; writes 0 for k > q so
// the AV GEMM needs no mask). One block of 256 per row; 4 elems/thread.
// ---------------------------------------------------------------------------
__global__ __launch_bounds__(256) void softmax_causal(float* __restrict__ scores) {
  const int row = blockIdx.x;        // bh*T + q
  const int q = row & (TT - 1);
  float* p = scores + (size_t)row * TT;
  const int tid = threadIdx.x;
  __shared__ float red[256];

  float r[4];
  float mx = -3.0e38f;
#pragma unroll
  for (int i = 0; i < 4; ++i) {
    const int k = tid + i * 256;
    const float v = (k <= q) ? p[k] : -3.0e38f;
    r[i] = v;
    mx = fmaxf(mx, v);
  }
  red[tid] = mx;
  __syncthreads();
  for (int s = 128; s > 0; s >>= 1) {
    if (tid < s) red[tid] = fmaxf(red[tid], red[tid + s]);
    __syncthreads();
  }
  mx = red[0];
  __syncthreads();

  float sum = 0.f;
#pragma unroll
  for (int i = 0; i < 4; ++i) {
    const int k = tid + i * 256;
    const float e = (k <= q) ? __expf(r[i] - mx) : 0.f;
    r[i] = e;
    sum += e;
  }
  red[tid] = sum;
  __syncthreads();
  for (int s = 128; s > 0; s >>= 1) {
    if (tid < s) red[tid] += red[tid + s];
    __syncthreads();
  }
  const float inv = 1.0f / red[0];
#pragma unroll
  for (int i = 0; i < 4; ++i) p[tid + i * 256] = r[i] * inv;
}

// ---------------------------------------------------------------------------
// AV: out[b,q,h*64+d] = sum_k P[bh,q,k] * V[b,k,h,d].  N=DH=64, K-dim = T.
// Block: 128 threads = 4 waves, tile 64(q) x 64(d). K-loop stops at the
// causal diagonal (P is exactly zero beyond it).
// ---------------------------------------------------------------------------
__global__ __launch_bounds__(128) void attn_av(
    const float* __restrict__ scores, const float* __restrict__ qkv,
    float* __restrict__ outp) {
  const int q0 = blockIdx.x * 64;
  const int bh = blockIdx.y;
  const int b = bh / HH, h = bh % HH;

  __shared__ __bf16 Ps[64][40];
  __shared__ __bf16 Vs[32][72];

  const int tid = threadIdx.x;
  const int lane = tid & 31, wv = tid >> 5;
  const int lm = lane & 15, hi = lane >> 4;
  const int rr = tid >> 1, cb = (tid & 1) * 16;     // P stage
  const int vr = tid >> 2, vc = (tid & 3) * 16;     // V stage

  v8f acc[4] = {};
  const int kmax = q0 + 64;

  for (int kk = 0; kk < kmax; kk += 32) {
    const float4* pp = reinterpret_cast<const float4*>(
        scores + ((size_t)bh * TT + q0 + rr) * TT + kk + cb);
    float4 p0 = pp[0], p1 = pp[1], p2 = pp[2], p3 = pp[3];
    float pt[16] = {p0.x, p0.y, p0.z, p0.w, p1.x, p1.y, p1.z, p1.w,
                    p2.x, p2.y, p2.z, p2.w, p3.x, p3.y, p3.z, p3.w};
#pragma unroll
    for (int j = 0; j < 16; ++j) Ps[rr][cb + j] = (__bf16)pt[j];

    const float4* vp = reinterpret_cast<const float4*>(
        qkv + (size_t)(b * TT + kk + vr) * (3 * DD) + 2 * DD + h * 64 + vc);
    float4 v0 = vp[0], v1 = vp[1], v2 = vp[2], v3 = vp[3];
    float vt[16] = {v0.x, v0.y, v0.z, v0.w, v1.x, v1.y, v1.z, v1.w,
                    v2.x, v2.y, v2.z, v2.w, v3.x, v3.y, v3.z, v3.w};
#pragma unroll
    for (int j = 0; j < 16; ++j) Vs[vr][vc + j] = (__bf16)vt[j];

    __syncthreads();

    v16bf af;
    const int arow = wv * 16 + lm;
#pragma unroll
    for (int j = 0; j < 16; ++j) af[j] = Ps[arow][16 * hi + j];
#pragma unroll
    for (int nn = 0; nn < 4; ++nn) {
      v16bf bf_;
      const int bcol = nn * 16 + lm;
#pragma unroll
      for (int j = 0; j < 16; ++j) bf_[j] = Vs[16 * hi + j][bcol];
      acc[nn] = __builtin_amdgcn_wmma_f32_16x16x32_bf16(
          false, af, false, bf_, (short)0, acc[nn], false, false);
    }
    __syncthreads();
  }

#pragma unroll
  for (int nn = 0; nn < 4; ++nn) {
    const int d = nn * 16 + lm;
#pragma unroll
    for (int r = 0; r < 8; ++r) {
      const int q = q0 + wv * 16 + hi * 8 + r;
      outp[(size_t)(b * TT + q) * DD + h * 64 + d] = acc[nn][r];
    }
  }
}

// ---------------------------------------------------------------------------
// LayerNorm over D=1024 per row; block 256, 4 elems/thread held in registers.
// ---------------------------------------------------------------------------
__global__ __launch_bounds__(256) void layernorm_k(
    const float* __restrict__ in, const float* __restrict__ g,
    const float* __restrict__ bta, float* __restrict__ outp) {
  const int row = blockIdx.x;
  const float* p = in + (size_t)row * DD;
  const int tid = threadIdx.x;
  __shared__ float red[256];

  float r[4];
  float s = 0.f;
#pragma unroll
  for (int i = 0; i < 4; ++i) {
    r[i] = p[tid + i * 256];
    s += r[i];
  }
  red[tid] = s;
  __syncthreads();
  for (int st = 128; st > 0; st >>= 1) {
    if (tid < st) red[tid] += red[tid + st];
    __syncthreads();
  }
  const float mean = red[0] * (1.0f / DD);
  __syncthreads();

  float vs = 0.f;
#pragma unroll
  for (int i = 0; i < 4; ++i) {
    const float d = r[i] - mean;
    vs += d * d;
  }
  red[tid] = vs;
  __syncthreads();
  for (int st = 128; st > 0; st >>= 1) {
    if (tid < st) red[tid] += red[tid + st];
    __syncthreads();
  }
  const float inv = rsqrtf(red[0] * (1.0f / DD) + 1e-5f);
#pragma unroll
  for (int i = 0; i < 4; ++i) {
    const int d = tid + i * 256;
    outp[(size_t)row * DD + d] = (r[i] - mean) * inv * g[d] + bta[d];
  }
}

// ---------------------------------------------------------------------------
// Elementwise helpers.
// ---------------------------------------------------------------------------
__global__ void embed_add_k(const int* __restrict__ x,
                            const float* __restrict__ embed,
                            const float* __restrict__ pos,
                            float* __restrict__ h) {
  const int row = blockIdx.x;       // b*T + t
  const int t = row & (TT - 1);
  const int tok = x[row];
  for (int d = threadIdx.x; d < DD; d += blockDim.x)
    h[(size_t)row * DD + d] =
        embed[(size_t)tok * DD + d] + pos[(size_t)t * DD + d];
}

__global__ void sub_k(const float* __restrict__ a, const float* __restrict__ b,
                      float* __restrict__ c) {
  const size_t base = (size_t)blockIdx.x * DD;
  for (int d = threadIdx.x; d < DD; d += blockDim.x)
    c[base + d] = a[base + d] - b[base + d];
}

__global__ void concat_k(const float* __restrict__ a, const float* __restrict__ b,
                         float* __restrict__ c) {
  const int row = blockIdx.x;
  for (int d = threadIdx.x; d < 2 * DD; d += blockDim.x) {
    const float v = (d < DD) ? a[(size_t)row * DD + d]
                             : b[(size_t)row * DD + (d - DD)];
    c[(size_t)row * (2 * DD) + d] = v;
  }
}

// out = pred + gate * corr
__global__ void combine_k(const float* __restrict__ pred,
                          const float* __restrict__ gate,
                          const float* __restrict__ corr,
                          float* __restrict__ outp) {
  const size_t base = (size_t)blockIdx.x * DD;
  for (int d = threadIdx.x; d < DD; d += blockDim.x)
    outp[base + d] = pred[base + d] + gate[base + d] * corr[base + d];
}

// ---------------------------------------------------------------------------
// Host orchestration.
// ---------------------------------------------------------------------------
extern "C" void kernel_launch(void* const* d_in, const int* in_sizes, int n_in,
                              void* d_out, int out_size, void* d_ws,
                              size_t ws_size, hipStream_t stream) {
  const int*   x     = (const int*)d_in[0];
  const float* embed = (const float*)d_in[1];
  const float* pos   = (const float*)d_in[2];
  const float* Wqkv  = (const float*)d_in[3];
  const float* bqkv  = (const float*)d_in[4];
  const float* Wo    = (const float*)d_in[5];
  const float* bo    = (const float*)d_in[6];
  const float* Wg    = (const float*)d_in[7];
  const float* bg    = (const float*)d_in[8];
  const float* ln1g  = (const float*)d_in[9];
  const float* ln1b  = (const float*)d_in[10];
  const float* W1    = (const float*)d_in[11];
  const float* b1    = (const float*)d_in[12];
  const float* W2    = (const float*)d_in[13];
  const float* b2    = (const float*)d_in[14];
  const float* ln2g  = (const float*)d_in[15];
  const float* ln2b  = (const float*)d_in[16];
  const float* lnfg  = (const float*)d_in[17];
  const float* lnfb  = (const float*)d_in[18];
  float* out = (float*)d_out;

  // Workspace layout (fp32), total ~265 MB.
  float* ws   = (float*)d_ws;
  float* h_   = ws;
  float* n1   = h_   + (size_t)MM * DD;
  float* qkv  = n1   + (size_t)MM * DD;
  float* sc   = qkv  + (size_t)MM * 3 * DD;
  float* pred = sc   + (size_t)BB * HH * TT * TT;
  float* dif  = pred + (size_t)MM * DD;
  float* corr = dif  + (size_t)MM * DD;
  float* catb = corr + (size_t)MM * DD;
  float* gate = catb + (size_t)MM * 2 * DD;
  float* aout = gate + (size_t)MM * DD;
  float* n2   = aout + (size_t)MM * DD;
  float* mid  = n2   + (size_t)MM * DD;

  embed_add_k<<<MM, 256, 0, stream>>>(x, embed, pos, h_);

  auto attend = [&](const float* a_in, const float* Wl, const float* bl,
                    float* o) {
    gemm_bf16_wmma<0><<<dim3(3 * DD / 128, MM / 128), 256, 0, stream>>>(
        a_in, DD, Wl, DD, bl, nullptr, qkv, 3 * DD, DD);
    attn_scores<<<dim3(TT / 64, TT / 64, BB * HH), 128, 0, stream>>>(qkv, sc);
    softmax_causal<<<BB * HH * TT, 256, 0, stream>>>(sc);
    attn_av<<<dim3(TT / 64, BB * HH), 128, 0, stream>>>(sc, qkv, o);
  };

  for (int l = 0; l < LLAY; ++l) {
    layernorm_k<<<MM, 256, 0, stream>>>(h_, ln1g + l * DD, ln1b + l * DD, n1);

    // predictor pass
    attend(n1, Wqkv + (size_t)(l * 2 + 0) * 3 * DD * DD,
           bqkv + (size_t)(l * 2 + 0) * 3 * DD, pred);
    // corrector pass on (n1 - pred)
    sub_k<<<MM, 256, 0, stream>>>(n1, pred, dif);
    attend(dif, Wqkv + (size_t)(l * 2 + 1) * 3 * DD * DD,
           bqkv + (size_t)(l * 2 + 1) * 3 * DD, corr);

    // gate = sigmoid(concat(pred, corr) @ Wg^T + bg)
    concat_k<<<MM, 256, 0, stream>>>(pred, corr, catb);
    gemm_bf16_wmma<2><<<dim3(DD / 128, MM / 128), 256, 0, stream>>>(
        catb, 2 * DD, Wg + (size_t)l * DD * 2 * DD, 2 * DD, bg + l * DD,
        nullptr, gate, DD, 2 * DD);
    combine_k<<<MM, 256, 0, stream>>>(pred, gate, corr, aout);

    // h += aout @ Wo^T + bo
    gemm_bf16_wmma<0><<<dim3(DD / 128, MM / 128), 256, 0, stream>>>(
        aout, DD, Wo + (size_t)l * DD * DD, DD, bo + l * DD, h_, h_, DD, DD);

    // FFN
    layernorm_k<<<MM, 256, 0, stream>>>(h_, ln2g + l * DD, ln2b + l * DD, n2);
    gemm_bf16_wmma<1><<<dim3(FFD / 128, MM / 128), 256, 0, stream>>>(
        n2, DD, W1 + (size_t)l * FFD * DD, DD, b1 + l * FFD, nullptr, mid, FFD,
        DD);
    gemm_bf16_wmma<0><<<dim3(DD / 128, MM / 128), 256, 0, stream>>>(
        mid, FFD, W2 + (size_t)l * DD * FFD, FFD, b2 + l * DD, h_, h_, DD, FFD);
  }

  // Final LN + logits = nf @ embed^T  (N = 32000, K = 1024)
  layernorm_k<<<MM, 256, 0, stream>>>(h_, lnfg, lnfb, n1);
  gemm_bf16_wmma<0><<<dim3(VV / 128, MM / 128), 256, 0, stream>>>(
      n1, DD, embed, DD, nullptr, nullptr, out, VV, DD);
}
